// _MAGBlock_7473243095584
// MI455X (gfx1250) — matrix operand order repaired and checked by reading, not compiled, not run on hardware
//
#include <hip/hip_runtime.h>
#include <cstdint>

// ---------------- problem constants ----------------
#define B_    4
#define T_    2048
#define D_    1024
#define DH_   2048
#define NP_   64
#define TA_   2112      // T + NP
#define WIN_  256
#define CH_   64
#define FD_   4096      // FM * D
#define NQT_  132       // TA/16
#define BAND_ 352       // 64 prefix + 288 window keys (11 * 32)
#define NC_   33        // TA / CHUNK
#define LDVT_ 2144      // TA + 32 pad for 32-wide K blocks
#define BTA_  (B_*TA_)
#define BT_   (B_*T_)

typedef __bf16 bf16;
typedef __attribute__((ext_vector_type(16))) __bf16 v16bf;
typedef __attribute__((ext_vector_type(8)))  float  v8f;
typedef int v4i_ __attribute__((vector_size(16)));
typedef __attribute__((address_space(1))) v4i_* gv4i_p;   // global int4*
typedef __attribute__((address_space(3))) v4i_* lv4i_p;   // LDS int4*

union FragU { uint4 q[2]; v16bf v; };

__device__ __forceinline__ float sigf(float x) { return 1.0f / (1.0f + __expf(-x)); }

#if (__has_builtin(__builtin_amdgcn_global_load_async_to_lds_b128) && \
     __has_builtin(__builtin_amdgcn_s_wait_asynccnt))
#define HAS_ASYNC_LDS 1
#else
#define HAS_ASYNC_LDS 0
#endif

// ---------------- epilogue ids ----------------
constexpr int EP_BF16 = 0;      // Cb = acc
constexpr int EP_F32 = 1;       // Cf = acc
constexpr int EP_RES = 2;       // Cf = acc + R
constexpr int EP_SILU = 3;      // Cb = silu(acc), optional P = acc (pre-act)
constexpr int EP_DPRED = 4;     // Cb = (acc - V) / CHUNK
constexpr int EP_DH = 5;        // Cb = acc * silu'(V)
constexpr int EP_GATE = 6;      // g = sig(acc+bias); Cb = g*V + (1-g)*V2
constexpr int EP_BIAS_SILU = 7; // Cb = silu(acc + bias)
constexpr int EP_RES_BIAS = 8;  // Cf = acc + bias + R
constexpr int EP_TRANS = 9;     // CbT[n*ldcT + m] = acc

struct GemmArgs {
  const bf16* A; const bf16* Bt;
  float* Cf; bf16* Cb; bf16* CbT;
  const float* R; const float* V; const float* V2;
  float* P; const float* bias;
  int M, N, K, lda, ldb, ldc, ldcT;
  long sA, sBt, sCf, sCb, sCbT, sR, sV, sP;
};

// C[M,N] = A[M,K] @ Bt[N,K]^T  (bf16 operands, f32 accumulate via WMMA).
// Block = 4 waves (2x2), block tile 64x64; each wave owns a 32x32 subtile
// computed as a 2x2 grid of 16x16 WMMA accumulators (A/B fragment reuse).
template<int EPI>
__global__ void __launch_bounds__(128) gemm_nt_k(GemmArgs g) {
  const int lane = threadIdx.x & 31;
  const int wave = threadIdx.x >> 5;
  const int wm = (wave >> 1) * 32;
  const int wn = (wave & 1) * 32;
  const int m0 = blockIdx.y * 64 + wm;
  const int n0 = blockIdx.x * 64 + wn;
  const int bz = blockIdx.z;
  if (blockIdx.y * 64 >= g.M || blockIdx.x * 64 >= g.N) return;
  const bf16* A  = g.A  + (size_t)bz * g.sA;
  const bf16* Bt = g.Bt + (size_t)bz * g.sBt;
  const int hi = (lane < 16) ? 0 : 1;
  const int ml = lane & 15;
  // A frag: lane holds row M, K-halves {kb..kb+7, kb+16..kb+23}, kb = 8*hi
  const bf16* ap0 = A + (size_t)(m0 + ml) * g.lda + (hi ? 8 : 0);
  const bf16* ap1 = ap0 + (size_t)16 * g.lda;
  // B frag: lane holds col N, K-halves {kb..kb+7, kb+16..kb+23}, kb = 8*hi
  const bf16* bp0 = Bt + (size_t)(n0 + ml) * g.ldb + (hi ? 8 : 0);
  const bf16* bp1 = bp0 + (size_t)16 * g.ldb;
  v8f acc00 = {}, acc01 = {}, acc10 = {}, acc11 = {};
  for (int k0 = 0; k0 < g.K; k0 += 32) {
    FragU a0, a1, b0, b1;
    a0.q[0] = *(const uint4*)(ap0 + k0);
    a0.q[1] = *(const uint4*)(ap0 + k0 + 16);
    a1.q[0] = *(const uint4*)(ap1 + k0);
    a1.q[1] = *(const uint4*)(ap1 + k0 + 16);
    b0.q[0] = *(const uint4*)(bp0 + k0);
    b0.q[1] = *(const uint4*)(bp0 + k0 + 16);
    b1.q[0] = *(const uint4*)(bp1 + k0);
    b1.q[1] = *(const uint4*)(bp1 + k0 + 16);
    __builtin_prefetch(ap0 + k0 + 256, 0, 0);   // global_prefetch_b8
    __builtin_prefetch(bp0 + k0 + 256, 0, 0);
    acc00 = __builtin_amdgcn_wmma_f32_16x16x32_bf16(false, a0.v, false, b0.v,
                                                    (short)0, acc00, false, false);
    acc01 = __builtin_amdgcn_wmma_f32_16x16x32_bf16(false, a0.v, false, b1.v,
                                                    (short)0, acc01, false, false);
    acc10 = __builtin_amdgcn_wmma_f32_16x16x32_bf16(false, a1.v, false, b0.v,
                                                    (short)0, acc10, false, false);
    acc11 = __builtin_amdgcn_wmma_f32_16x16x32_bf16(false, a1.v, false, b1.v,
                                                    (short)0, acc11, false, false);
  }
  v8f accs[2][2] = {{acc00, acc01}, {acc10, acc11}};
#pragma unroll
  for (int mi = 0; mi < 2; ++mi) {
#pragma unroll
    for (int ni = 0; ni < 2; ++ni) {
      const int n = n0 + ni * 16 + ml;
#pragma unroll
      for (int r = 0; r < 8; ++r) {
        const int m = m0 + mi * 16 + r + hi * 8;
        const float av = accs[mi][ni][r];
        const size_t ci = (size_t)m * g.ldc + n;
        if constexpr (EPI == EP_BF16) {
          g.Cb[(size_t)bz * g.sCb + ci] = (bf16)av;
        } else if constexpr (EPI == EP_F32) {
          g.Cf[(size_t)bz * g.sCf + ci] = av;
        } else if constexpr (EPI == EP_RES) {
          g.Cf[(size_t)bz * g.sCf + ci] = av + g.R[(size_t)bz * g.sR + ci];
        } else if constexpr (EPI == EP_SILU) {
          if (g.P) g.P[(size_t)bz * g.sP + ci] = av;
          g.Cb[(size_t)bz * g.sCb + ci] = (bf16)(av * sigf(av));
        } else if constexpr (EPI == EP_DPRED) {
          g.Cb[(size_t)bz * g.sCb + ci] =
              (bf16)((av - g.V[(size_t)bz * g.sV + ci]) * (1.0f / (float)CH_));
        } else if constexpr (EPI == EP_DH) {
          const float pre = g.V[(size_t)bz * g.sV + ci];
          const float s = sigf(pre);
          g.Cb[(size_t)bz * g.sCb + ci] =
              (bf16)(av * (s * (1.0f + pre * (1.0f - s))));
        } else if constexpr (EPI == EP_GATE) {
          const float gg = sigf(av + g.bias[n]);
          g.Cb[ci] = (bf16)(gg * g.V[ci] + (1.0f - gg) * g.V2[ci]);
        } else if constexpr (EPI == EP_BIAS_SILU) {
          const float t = av + g.bias[n];
          g.Cb[(size_t)bz * g.sCb + ci] = (bf16)(t * sigf(t));
        } else if constexpr (EPI == EP_RES_BIAS) {
          g.Cf[(size_t)bz * g.sCf + ci] =
              av + g.bias[n] + g.R[(size_t)bz * g.sR + ci];
        } else if constexpr (EPI == EP_TRANS) {
          g.CbT[(size_t)bz * g.sCbT + (size_t)n * g.ldcT + m] = (bf16)av;
        }
      }
    }
  }
}

// G[M,N] = A[K,M]^T @ B[K,N] (K = 64), fused momentum update:
// S = eta*S - theta*G; W = (1-alpha)*W + S; emit f32 W + bf16 (opt) + bf16-T.
// A/B tiles staged through LDS (async global->LDS when toolchain supports it).
struct TnArgs {
  const bf16* A; const bf16* Bm;
  float* S; float* Wf; bf16* Wb; bf16* WbT;
  const float* theta; const float* eta; const float* alpha;
  int M, N, K, lda, ldb, ldw, ldwt;
  long sA, sB, sW;
};

__global__ void __launch_bounds__(128) gemm_tn_update_k(TnArgs g) {
  __shared__ __align__(16) bf16 As[64][80];   // [K][Mtile], padded stride
  __shared__ __align__(16) bf16 Bs[64][80];   // [K][Ntile]
  const int lane = threadIdx.x & 31;
  const int wave = threadIdx.x >> 5;
  const int mb = blockIdx.y * 64;
  const int nb = blockIdx.x * 64;
  const int bz = blockIdx.z;
  const bf16* A  = g.A  + (size_t)bz * g.sA;
  const bf16* Bm = g.Bm + (size_t)bz * g.sB;
  // stage 64x64 A (cols mb..mb+63) and B (cols nb..nb+63): 512 16B chunks each
  for (int i = 0; i < 4; ++i) {
    const int c = threadIdx.x + i * 128;
    const int row = c >> 3;
    const int col = (c & 7) * 8;
    const bf16* gA = A  + (size_t)row * g.lda + mb + col;
    const bf16* gB = Bm + (size_t)row * g.ldb + nb + col;
#if HAS_ASYNC_LDS
    __builtin_amdgcn_global_load_async_to_lds_b128(
        (gv4i_p)gA, (lv4i_p)&As[row][col], 0, 0);
    __builtin_amdgcn_global_load_async_to_lds_b128(
        (gv4i_p)gB, (lv4i_p)&Bs[row][col], 0, 0);
#else
    *(uint4*)&As[row][col] = *(const uint4*)gA;
    *(uint4*)&Bs[row][col] = *(const uint4*)gB;
#endif
  }
#if HAS_ASYNC_LDS
  __builtin_amdgcn_s_wait_asynccnt(0);
#endif
  __syncthreads();

  const int wm = (wave >> 1) * 32;
  const int wn = (wave & 1) * 32;
  const int hi = (lane < 16) ? 0 : 1;
  const int ml = lane & 15;
  const int kbA = hi ? 16 : 0;
  const int kbB = hi ? 16 : 0;
  v8f acc00 = {}, acc01 = {}, acc10 = {}, acc11 = {};
#pragma unroll
  for (int k0 = 0; k0 < 64; k0 += 32) {
    v16bf a0, a1, b0, b1;
#pragma unroll
    for (int e = 0; e < 8; ++e) {
      a0[e]     = As[k0 + kbA + e][wm + ml];
      a0[e + 8] = As[k0 + kbA + e + 8][wm + ml];
      a1[e]     = As[k0 + kbA + e][wm + 16 + ml];
      a1[e + 8] = As[k0 + kbA + e + 8][wm + 16 + ml];
    }
#pragma unroll
    for (int e = 0; e < 16; ++e) {
      b0[e] = Bs[k0 + kbB + e][wn + ml];
      b1[e] = Bs[k0 + kbB + e][wn + 16 + ml];
    }
    acc00 = __builtin_amdgcn_wmma_f32_16x16x32_bf16(false, a0, false, b0,
                                                    (short)0, acc00, false, false);
    acc01 = __builtin_amdgcn_wmma_f32_16x16x32_bf16(false, a0, false, b1,
                                                    (short)0, acc01, false, false);
    acc10 = __builtin_amdgcn_wmma_f32_16x16x32_bf16(false, a1, false, b0,
                                                    (short)0, acc10, false, false);
    acc11 = __builtin_amdgcn_wmma_f32_16x16x32_bf16(false, a1, false, b1,
                                                    (short)0, acc11, false, false);
  }
  v8f accs[2][2] = {{acc00, acc01}, {acc10, acc11}};
  const float th = *g.theta, et = *g.eta, al = *g.alpha;
  float* S  = g.S  + (size_t)bz * g.sW;
  float* Wf = g.Wf + (size_t)bz * g.sW;
  bf16* Wb  = g.Wb ? (g.Wb + (size_t)bz * g.sW) : nullptr;
  bf16* WbT = g.WbT + (size_t)bz * g.sW;
#pragma unroll
  for (int mi = 0; mi < 2; ++mi) {
#pragma unroll
    for (int ni = 0; ni < 2; ++ni) {
      const int n = nb + wn + ni * 16 + ml;
#pragma unroll
      for (int r = 0; r < 8; ++r) {
        const int m = mb + wm + mi * 16 + r + hi * 8;
        const size_t wi = (size_t)m * g.ldw + n;
        const float sv = et * S[wi] - th * accs[mi][ni][r];
        S[wi] = sv;
        const float wv = (1.0f - al) * Wf[wi] + sv;
        Wf[wi] = wv;
        if (Wb) Wb[wi] = (bf16)wv;
        WbT[(size_t)n * g.ldwt + m] = (bf16)wv;
      }
    }
  }
}

// ---------------- attention: banded scores via WMMA ----------------
__global__ void __launch_bounds__(32) attn_score_k(const bf16* qb, const bf16* kb,
                                                   float* band) {
  const int lane = threadIdx.x;
  const int ct = blockIdx.x, qt = blockIdx.y, b = blockIdx.z;
  const int c0 = ct * 16;
  const int wstart = qt * 16 - WIN_;
  const int hi = (lane < 16) ? 0 : 1;
  const int jl = lane & 15;
  const int j0 = (c0 < NP_) ? c0 : (wstart + (c0 - NP_));
  int jr = j0 + jl;
  jr = jr < 0 ? 0 : (jr > TA_ - 1 ? TA_ - 1 : jr);
  const bf16* Q  = qb + (size_t)b * TA_ * D_;
  const bf16* Km = kb + (size_t)b * TA_ * D_;
  const bf16* ap = Q  + (size_t)(qt * 16 + jl) * D_ + (hi ? 8 : 0);
  const bf16* bp = Km + (size_t)jr * D_ + (hi ? 8 : 0);
  v8f acc = {};
  for (int k0 = 0; k0 < D_; k0 += 32) {
    FragU a, bv;
    a.q[0]  = *(const uint4*)(ap + k0);
    a.q[1]  = *(const uint4*)(ap + k0 + 16);
    bv.q[0] = *(const uint4*)(bp + k0);
    bv.q[1] = *(const uint4*)(bp + k0 + 16);
    acc = __builtin_amdgcn_wmma_f32_16x16x32_bf16(false, a.v, false, bv.v,
                                                  (short)0, acc, false, false);
  }
  float* brow = band + (size_t)(b * NQT_ + qt) * 16 * BAND_;
  const int c = c0 + jl;
  const int j = (c < NP_) ? c : (wstart + (c - NP_));
#pragma unroll
  for (int r = 0; r < 8; ++r) {
    const int i = qt * 16 + r + hi * 8;
    const bool ok = (c < NP_) ? (j <= i)
                              : ((j >= NP_) && (j <= i) && (i - j < WIN_));
    brow[(size_t)(r + hi * 8) * BAND_ + c] = ok ? acc[r] : -1e30f;
  }
}

__global__ void __launch_bounds__(128) attn_softmax_k(const float* band, bf16* Pb) {
  __shared__ float red[128];
  const int row = blockIdx.x;
  const float* s = band + (size_t)row * BAND_;
  bf16* p = Pb + (size_t)row * BAND_;
  float mx = -3.4e38f;
  for (int i = threadIdx.x; i < BAND_; i += 128) mx = fmaxf(mx, s[i]);
  red[threadIdx.x] = mx; __syncthreads();
  for (int o = 64; o > 0; o >>= 1) {
    if (threadIdx.x < o) red[threadIdx.x] = fmaxf(red[threadIdx.x], red[threadIdx.x + o]);
    __syncthreads();
  }
  mx = red[0]; __syncthreads();
  float sum = 0.0f;
  for (int i = threadIdx.x; i < BAND_; i += 128) sum += __expf(s[i] - mx);
  red[threadIdx.x] = sum; __syncthreads();
  for (int o = 64; o > 0; o >>= 1) {
    if (threadIdx.x < o) red[threadIdx.x] += red[threadIdx.x + o];
    __syncthreads();
  }
  const float inv = 1.0f / red[0];
  for (int i = threadIdx.x; i < BAND_; i += 128) p[i] = (bf16)(__expf(s[i] - mx) * inv);
}

// out[16,D] = P[16,BAND] @ V(gathered band rows) using transposed V (bf16, ld LDVT_)
__global__ void __launch_bounds__(128) attn_av_k(const bf16* Pb, const bf16* vT,
                                                 bf16* ao) {
  const int lane = threadIdx.x & 31, wave = threadIdx.x >> 5;
  const int n0 = blockIdx.x * 64 + wave * 16;
  const int qt = blockIdx.y, b = blockIdx.z;
  const int hi = (lane < 16) ? 0 : 1;
  const int nl = lane & 15;
  const bf16* Prow = Pb + (size_t)(b * NQT_ + qt) * 16 * BAND_;
  const bf16* VT   = vT + (size_t)b * D_ * LDVT_;
  const bf16* ap   = Prow + (size_t)nl * BAND_ + (hi ? 8 : 0);
  const bf16* vrow = VT + (size_t)(n0 + nl) * LDVT_ + (hi ? 8 : 0);
  const int wstart = qt * 16 - WIN_;
  v8f acc = {};
  for (int kk = 0; kk < BAND_; kk += 32) {
    const int keyb = (kk < NP_) ? kk : (wstart + (kk - NP_));
    if (kk >= NP_ && keyb + 31 < NP_) continue;  // fully masked block (also keyb<0)
    FragU a, bv;
    a.q[0]  = *(const uint4*)(ap + kk);
    a.q[1]  = *(const uint4*)(ap + kk + 16);
    const bf16* bp = vrow + keyb;
    bv.q[0] = *(const uint4*)(bp);
    bv.q[1] = *(const uint4*)(bp + 16);
    acc = __builtin_amdgcn_wmma_f32_16x16x32_bf16(false, a.v, false, bv.v,
                                                  (short)0, acc, false, false);
  }
#pragma unroll
  for (int r = 0; r < 8; ++r)
    ao[((size_t)b * TA_ + qt * 16 + r + hi * 8) * D_ + n0 + nl] = (bf16)acc[r];
}

// ---------------- row kernels: LN / L2-normalize ----------------
__global__ void __launch_bounds__(256) ln_rows_k(const float* src, const float* gm,
                                                 const float* bt, bf16* dst) {
  __shared__ float red[256];
  const int row = blockIdx.x;
  const float* s = src + (size_t)row * D_;
  float lsum = 0.0f;
  for (int i = threadIdx.x; i < D_; i += 256) lsum += s[i];
  red[threadIdx.x] = lsum; __syncthreads();
  for (int o = 128; o > 0; o >>= 1) {
    if (threadIdx.x < o) red[threadIdx.x] += red[threadIdx.x + o];
    __syncthreads();
  }
  const float mean = red[0] / (float)D_; __syncthreads();
  float lss = 0.0f;
  for (int i = threadIdx.x; i < D_; i += 256) { const float d = s[i] - mean; lss += d * d; }
  red[threadIdx.x] = lss; __syncthreads();
  for (int o = 128; o > 0; o >>= 1) {
    if (threadIdx.x < o) red[threadIdx.x] += red[threadIdx.x + o];
    __syncthreads();
  }
  const float inv = rsqrtf(red[0] / (float)D_ + 1e-5f);
  for (int i = threadIdx.x; i < D_; i += 256)
    dst[(size_t)row * D_ + i] = (bf16)((s[i] - mean) * inv * gm[i] + bt[i]);
}

__global__ void __launch_bounds__(256) ln_out1_k(const float* comb, const float* x,
                                                 const float* gm, const float* bt,
                                                 float* out1) {
  __shared__ float red[256];
  const int row = blockIdx.x;                 // 0..BT-1
  const int b = row / T_, t = row % T_;
  const float* s = comb + (size_t)(b * TA_ + NP_ + t) * D_;
  const float* xr = x + (size_t)row * D_;
  float* o = out1 + (size_t)row * D_;
  float lsum = 0.0f;
  for (int i = threadIdx.x; i < D_; i += 256) lsum += s[i];
  red[threadIdx.x] = lsum; __syncthreads();
  for (int oo = 128; oo > 0; oo >>= 1) {
    if (threadIdx.x < oo) red[threadIdx.x] += red[threadIdx.x + oo];
    __syncthreads();
  }
  const float mean = red[0] / (float)D_; __syncthreads();
  float lss = 0.0f;
  for (int i = threadIdx.x; i < D_; i += 256) { const float d = s[i] - mean; lss += d * d; }
  red[threadIdx.x] = lss; __syncthreads();
  for (int oo = 128; oo > 0; oo >>= 1) {
    if (threadIdx.x < oo) red[threadIdx.x] += red[threadIdx.x + oo];
    __syncthreads();
  }
  const float inv = rsqrtf(red[0] / (float)D_ + 1e-5f);
  for (int i = threadIdx.x; i < D_; i += 256)
    o[i] = xr[i] + (s[i] - mean) * inv * gm[i] + bt[i];
}

__global__ void __launch_bounds__(256) l2n_rows_k(const float* src, bf16* dst,
                                                  float scale) {
  __shared__ float red[256];
  const int row = blockIdx.x;
  const float* s = src + (size_t)row * D_;
  float lss = 0.0f;
  for (int i = threadIdx.x; i < D_; i += 256) lss += s[i] * s[i];
  red[threadIdx.x] = lss; __syncthreads();
  for (int o = 128; o > 0; o >>= 1) {
    if (threadIdx.x < o) red[threadIdx.x] += red[threadIdx.x + o];
    __syncthreads();
  }
  const float f = scale / fmaxf(sqrtf(red[0]), 1e-12f);
  for (int i = threadIdx.x; i < D_; i += 256)
    dst[(size_t)row * D_ + i] = (bf16)(s[i] * f);
}

// ---------------- elementwise helpers ----------------
__global__ void build_xaug_k(const float* x, const float* Pm, float* xa, long n) {
  for (long i = (long)blockIdx.x * blockDim.x + threadIdx.x; i < n;
       i += (long)gridDim.x * blockDim.x) {
    const long d = i % D_;
    const long row = (i / D_) % TA_;
    const long b = i / ((long)TA_ * D_);
    xa[i] = (row < NP_) ? Pm[row * D_ + d]
                        : x[((size_t)b * T_ + (row - NP_)) * D_ + d];
  }
}

__global__ void transconv_rep_k(const float* W, bf16* WT, long K, long N, long total) {
  const long kn = K * N;
  for (long i = (long)blockIdx.x * blockDim.x + threadIdx.x; i < total;
       i += (long)gridDim.x * blockDim.x) {
    const long b = i / kn, r = i % kn;
    const long k = r / N, nn = r % N;
    WT[(size_t)b * kn + nn * K + k] = (bf16)W[r];
  }
}

__global__ void copy_rep_k(const float* s, float* d, long per, long total) {
  for (long i = (long)blockIdx.x * blockDim.x + threadIdx.x; i < total;
       i += (long)gridDim.x * blockDim.x) d[i] = s[i % per];
}

__global__ void conv_rep_k(const float* s, bf16* d, long per, long total) {
  for (long i = (long)blockIdx.x * blockDim.x + threadIdx.x; i < total;
       i += (long)gridDim.x * blockDim.x) d[i] = (bf16)s[i % per];
}

__global__ void zero_k(float* p, long n) {
  for (long i = (long)blockIdx.x * blockDim.x + threadIdx.x; i < n;
       i += (long)gridDim.x * blockDim.x) p[i] = 0.0f;
}

__global__ void scale_comb_k(const float* yat, const float* ymm, const float* sa,
                             const float* sm, float* yaf, bf16* yab, float* ymf,
                             long n) {
  for (long i = (long)blockIdx.x * blockDim.x + threadIdx.x; i < n;
       i += (long)gridDim.x * blockDim.x) {
    const long d = i % D_;
    const float a = yat[i] * sa[d];
    yaf[i] = a;
    yab[i] = (bf16)a;
    ymf[i] = ymm[i] * sm[d];
  }
}

// ---------------- host orchestration ----------------
extern "C" void kernel_launch(void* const* d_in, const int* in_sizes, int n_in,
                              void* d_out, int out_size, void* d_ws, size_t ws_size,
                              hipStream_t stream) {
  (void)in_sizes; (void)n_in; (void)out_size; (void)ws_size;
  const float* x       = (const float*)d_in[0];
  const float* Pm      = (const float*)d_in[1];
  const float* g_attn  = (const float*)d_in[2];
  const float* b_attn  = (const float*)d_in[3];
  const float* Wq      = (const float*)d_in[4];
  const float* Wk      = (const float*)d_in[5];
  const float* Wv      = (const float*)d_in[6];
  const float* Wo      = (const float*)d_in[7];
  const float* g_mem   = (const float*)d_in[8];
  const float* b_mem   = (const float*)d_in[9];
  const float* Wk_m    = (const float*)d_in[10];
  const float* Wv_m    = (const float*)d_in[11];
  const float* Wq_m    = (const float*)d_in[12];
  const float* M1      = (const float*)d_in[13];
  const float* M2      = (const float*)d_in[14];
  const float* theta   = (const float*)d_in[15];
  const float* eta     = (const float*)d_in[16];
  const float* alpha   = (const float*)d_in[17];
  const float* sc_attn = (const float*)d_in[18];
  const float* sc_mem  = (const float*)d_in[19];
  const float* Wg      = (const float*)d_in[20];
  const float* bg      = (const float*)d_in[21];
  const float* Wout    = (const float*)d_in[22];
  const float* g_out   = (const float*)d_in[23];
  const float* b_out   = (const float*)d_in[24];
  const float* g_ffn   = (const float*)d_in[25];
  const float* b_ffn   = (const float*)d_in[26];
  const float* Wf1     = (const float*)d_in[27];
  const float* bf1     = (const float*)d_in[28];
  const float* Wf2     = (const float*)d_in[29];
  const float* bf2     = (const float*)d_in[30];
  float* out = (float*)d_out;

  char* ws = (char*)d_ws;
  size_t off = 0;
  auto alloc = [&](size_t bytes) -> void* {
    void* p = ws + off;
    off = (off + bytes + 255) & ~(size_t)255;
    return p;
  };
  const size_t NBD = (size_t)B_ * TA_ * D_;

  float* xaug  = (float*)alloc(NBD * 4);
  bf16* hln    = (bf16*)alloc(NBD * 2);
  float* raw   = (float*)alloc(NBD * 4);
  bf16* qb     = (bf16*)alloc(NBD * 2);
  bf16* kbuf   = (bf16*)alloc(NBD * 2);
  bf16* vT     = (bf16*)alloc((size_t)B_ * D_ * LDVT_ * 2);
  float* band  = (float*)alloc((size_t)B_ * NQT_ * 16 * BAND_ * 4);
  bf16* Pband  = (bf16*)alloc((size_t)B_ * NQT_ * 16 * BAND_ * 2);
  bf16* aout   = (bf16*)alloc(NBD * 2);
  float* yatt  = (float*)alloc(NBD * 4);
  bf16* km     = (bf16*)alloc(NBD * 2);
  float* vm    = (float*)alloc(NBD * 4);
  bf16* qm     = (bf16*)alloc(NBD * 2);
  float* ymem  = (float*)alloc(NBD * 4);
  // fast-weight state (per batch)
  const size_t WSZ = (size_t)D_ * DH_;
  float* W1f  = (float*)alloc((size_t)B_ * WSZ * 4);
  bf16* W1bT  = (bf16*)alloc((size_t)B_ * WSZ * 2);
  float* S1   = (float*)alloc((size_t)B_ * WSZ * 4);
  float* W2f  = (float*)alloc((size_t)B_ * WSZ * 4);
  bf16* W2b   = (bf16*)alloc((size_t)B_ * WSZ * 2);
  bf16* W2bT  = (bf16*)alloc((size_t)B_ * WSZ * 2);
  float* S2   = (float*)alloc((size_t)B_ * WSZ * 4);
  float* Hpre = (float*)alloc((size_t)B_ * CH_ * DH_ * 4);
  bf16* Hb    = (bf16*)alloc((size_t)B_ * CH_ * DH_ * 2);
  bf16* dPb   = (bf16*)alloc((size_t)B_ * CH_ * D_ * 2);
  bf16* dHb   = (bf16*)alloc((size_t)B_ * CH_ * DH_ * 2);
  bf16* Hqb   = (bf16*)alloc((size_t)B_ * CH_ * DH_ * 2);
  // combine + FFN
  float* yaf  = (float*)alloc(NBD * 4);
  bf16* yab   = (bf16*)alloc(NBD * 2);
  float* ymf  = (float*)alloc(NBD * 4);
  bf16* combB = (bf16*)alloc(NBD * 2);
  float* combF= (float*)alloc(NBD * 4);
  float* out1 = (float*)alloc((size_t)BT_ * D_ * 4);
  bf16* tln   = (bf16*)alloc((size_t)BT_ * D_ * 2);
  bf16* ub    = (bf16*)alloc((size_t)BT_ * FD_ * 2);
  // bf16-transposed weights ([N][K])
  bf16* WqT   = (bf16*)alloc((size_t)D_ * D_ * 2);
  bf16* WkT   = (bf16*)alloc((size_t)D_ * D_ * 2);
  bf16* WvT   = (bf16*)alloc((size_t)D_ * D_ * 2);
  bf16* WoT   = (bf16*)alloc((size_t)D_ * D_ * 2);
  bf16* WkmT  = (bf16*)alloc((size_t)D_ * D_ * 2);
  bf16* WvmT  = (bf16*)alloc((size_t)D_ * D_ * 2);
  bf16* WqmT  = (bf16*)alloc((size_t)D_ * D_ * 2);
  bf16* WgT   = (bf16*)alloc((size_t)D_ * D_ * 2);
  bf16* WoutT = (bf16*)alloc((size_t)D_ * D_ * 2);
  bf16* Wf1T  = (bf16*)alloc((size_t)D_ * FD_ * 2);
  bf16* Wf2T  = (bf16*)alloc((size_t)FD_ * D_ * 2);

  const int EB = 256, EG = 4096;
  // ---- prep: weight transposes + state init + x_aug ----
  transconv_rep_k<<<EG, EB, 0, stream>>>(Wq,   WqT,   D_, D_, (long)D_ * D_);
  transconv_rep_k<<<EG, EB, 0, stream>>>(Wk,   WkT,   D_, D_, (long)D_ * D_);
  transconv_rep_k<<<EG, EB, 0, stream>>>(Wv,   WvT,   D_, D_, (long)D_ * D_);
  transconv_rep_k<<<EG, EB, 0, stream>>>(Wo,   WoT,   D_, D_, (long)D_ * D_);
  transconv_rep_k<<<EG, EB, 0, stream>>>(Wk_m, WkmT,  D_, D_, (long)D_ * D_);
  transconv_rep_k<<<EG, EB, 0, stream>>>(Wv_m, WvmT,  D_, D_, (long)D_ * D_);
  transconv_rep_k<<<EG, EB, 0, stream>>>(Wq_m, WqmT,  D_, D_, (long)D_ * D_);
  transconv_rep_k<<<EG, EB, 0, stream>>>(Wg,   WgT,   D_, D_, (long)D_ * D_);
  transconv_rep_k<<<EG, EB, 0, stream>>>(Wout, WoutT, D_, D_, (long)D_ * D_);
  transconv_rep_k<<<EG, EB, 0, stream>>>(Wf1,  Wf1T,  D_, FD_, (long)D_ * FD_);
  transconv_rep_k<<<EG, EB, 0, stream>>>(Wf2,  Wf2T,  FD_, D_, (long)FD_ * D_);
  copy_rep_k<<<EG, EB, 0, stream>>>(M1, W1f, (long)WSZ, (long)B_ * WSZ);
  transconv_rep_k<<<EG, EB, 0, stream>>>(M1, W1bT, D_, DH_, (long)B_ * WSZ);
  copy_rep_k<<<EG, EB, 0, stream>>>(M2, W2f, (long)WSZ, (long)B_ * WSZ);
  conv_rep_k<<<EG, EB, 0, stream>>>(M2, W2b, (long)WSZ, (long)B_ * WSZ);
  transconv_rep_k<<<EG, EB, 0, stream>>>(M2, W2bT, DH_, D_, (long)B_ * WSZ);
  zero_k<<<EG, EB, 0, stream>>>(S1, (long)B_ * WSZ);
  zero_k<<<EG, EB, 0, stream>>>(S2, (long)B_ * WSZ);
  build_xaug_k<<<EG, EB, 0, stream>>>(x, Pm, xaug, (long)NBD);

  auto G = [](const bf16* A, const bf16* Bt, int M, int N, int K,
              int lda, int ldb, int ldc) {
    GemmArgs g{};
    g.A = A; g.Bt = Bt; g.M = M; g.N = N; g.K = K;
    g.lda = lda; g.ldb = ldb; g.ldc = ldc;
    return g;
  };

  // ---- attention branch ----
  ln_rows_k<<<BTA_, 256, 0, stream>>>(xaug, g_attn, b_attn, hln);
  { GemmArgs g = G(hln, WqT, BTA_, D_, D_, D_, D_, D_); g.Cf = raw;
    gemm_nt_k<EP_F32><<<dim3(D_/64, BTA_/64, 1), 128, 0, stream>>>(g); }
  l2n_rows_k<<<BTA_, 256, 0, stream>>>(raw, qb, 32.0f);  // sqrt(D)=32
  { GemmArgs g = G(hln, WkT, BTA_, D_, D_, D_, D_, D_); g.Cf = raw;
    gemm_nt_k<EP_F32><<<dim3(D_/64, BTA_/64, 1), 128, 0, stream>>>(g); }
  l2n_rows_k<<<BTA_, 256, 0, stream>>>(raw, kbuf, 1.0f);
  { GemmArgs g = G(hln, WvT, TA_, D_, D_, D_, D_, D_);
    g.CbT = vT; g.ldcT = LDVT_; g.sA = (long)TA_ * D_; g.sCbT = (long)D_ * LDVT_;
    gemm_nt_k<EP_TRANS><<<dim3(D_/64, TA_/64, B_), 128, 0, stream>>>(g); }
  attn_score_k<<<dim3(BAND_/16, NQT_, B_), 32, 0, stream>>>(qb, kbuf, band);
  attn_softmax_k<<<B_ * NQT_ * 16, 128, 0, stream>>>(band, Pband);
  attn_av_k<<<dim3(D_/64, NQT_, B_), 128, 0, stream>>>(Pband, vT, aout);
  { GemmArgs g = G(aout, WoT, BTA_, D_, D_, D_, D_, D_); g.Cf = yatt; g.R = xaug;
    gemm_nt_k<EP_RES><<<dim3(D_/64, BTA_/64, 1), 128, 0, stream>>>(g); }

  // ---- memory branch projections ----
  ln_rows_k<<<BTA_, 256, 0, stream>>>(xaug, g_mem, b_mem, hln);
  { GemmArgs g = G(hln, WkmT, BTA_, D_, D_, D_, D_, D_); g.Cb = km;
    gemm_nt_k<EP_BF16><<<dim3(D_/64, BTA_/64, 1), 128, 0, stream>>>(g); }
  { GemmArgs g = G(hln, WvmT, BTA_, D_, D_, D_, D_, D_); g.Cf = vm;
    gemm_nt_k<EP_F32><<<dim3(D_/64, BTA_/64, 1), 128, 0, stream>>>(g); }
  { GemmArgs g = G(hln, WqmT, BTA_, D_, D_, D_, D_, D_); g.Cb = qm;
    gemm_nt_k<EP_BF16><<<dim3(D_/64, BTA_/64, 1), 128, 0, stream>>>(g); }

  // ---- sequential fast-weight scan (33 chunk steps, batched over B) ----
  for (int s = 0; s < NC_; ++s) {
    const long co = (long)s * CH_ * D_;
    // k1: H = silu(Kc @ W1)  (pre-act kept for silu')
    { GemmArgs g = G(km + co, W1bT, CH_, DH_, D_, D_, D_, DH_);
      g.Cb = Hb; g.P = Hpre;
      g.sA = (long)TA_ * D_; g.sBt = (long)WSZ;
      g.sCb = (long)CH_ * DH_; g.sP = (long)CH_ * DH_;
      gemm_nt_k<EP_SILU><<<dim3(DH_/64, 1, B_), 128, 0, stream>>>(g); }
    // k2: dP = (H @ W2 - Vc) / CHUNK
    { GemmArgs g = G(Hb, W2bT, CH_, D_, DH_, DH_, DH_, D_);
      g.Cb = dPb; g.V = vm + co;
      g.sA = (long)CH_ * DH_; g.sBt = (long)WSZ;
      g.sCb = (long)CH_ * D_; g.sV = (long)TA_ * D_;
      gemm_nt_k<EP_DPRED><<<dim3(D_/64, 1, B_), 128, 0, stream>>>(g); }
    // k3: dH = (dP @ W2^T) * silu'(Hpre)
    { GemmArgs g = G(dPb, W2b, CH_, DH_, D_, D_, D_, DH_);
      g.Cb = dHb; g.V = Hpre;
      g.sA = (long)CH_ * D_; g.sBt = (long)WSZ;
      g.sCb = (long)CH_ * DH_; g.sV = (long)CH_ * DH_;
      gemm_nt_k<EP_DH><<<dim3(DH_/64, 1, B_), 128, 0, stream>>>(g); }
    // k4: g2 = H^T @ dP ; update S2/W2 (+bf16 copies)
    { TnArgs t{};
      t.A = Hb; t.Bm = dPb; t.S = S2; t.Wf = W2f; t.Wb = W2b; t.WbT = W2bT;
      t.theta = theta; t.eta = eta; t.alpha = alpha;
      t.M = DH_; t.N = D_; t.K = CH_; t.lda = DH_; t.ldb = D_;
      t.ldw = D_; t.ldwt = DH_;
      t.sA = (long)CH_ * DH_; t.sB = (long)CH_ * D_; t.sW = (long)WSZ;
      gemm_tn_update_k<<<dim3(D_/64, DH_/64, B_), 128, 0, stream>>>(t); }
    // k5: g1 = Kc^T @ dH ; update S1/W1 (+bf16-transposed copy)
    { TnArgs t{};
      t.A = km + co; t.Bm = dHb; t.S = S1; t.Wf = W1f; t.Wb = nullptr; t.WbT = W1bT;
      t.theta = theta; t.eta = eta; t.alpha = alpha;
      t.M = D_; t.N = DH_; t.K = CH_; t.lda = D_; t.ldb = DH_;
      t.ldw = DH_; t.ldwt = D_;
      t.sA = (long)TA_ * D_; t.sB = (long)CH_ * DH_; t.sW = (long)WSZ;
      gemm_tn_update_k<<<dim3(DH_/64, D_/64, B_), 128, 0, stream>>>(t); }
    // k6: Hq = silu(Qc @ W1new)
    { GemmArgs g = G(qm + co, W1bT, CH_, DH_, D_, D_, D_, DH_);
      g.Cb = Hqb;
      g.sA = (long)TA_ * D_; g.sBt = (long)WSZ; g.sCb = (long)CH_ * DH_;
      gemm_nt_k<EP_SILU><<<dim3(DH_/64, 1, B_), 128, 0, stream>>>(g); }
    // k7: y = Hq @ W2new
    { GemmArgs g = G(Hqb, W2bT, CH_, D_, DH_, DH_, DH_, D_);
      g.Cf = ymem + co;
      g.sA = (long)CH_ * DH_; g.sBt = (long)WSZ; g.sCf = (long)TA_ * D_;
      gemm_nt_k<EP_F32><<<dim3(D_/64, 1, B_), 128, 0, stream>>>(g); }
  }

  // ---- gated combine ----
  scale_comb_k<<<EG, EB, 0, stream>>>(yatt, ymem, sc_attn, sc_mem,
                                      yaf, yab, ymf, (long)NBD);
  { GemmArgs g = G(yab, WgT, BTA_, D_, D_, D_, D_, D_);
    g.Cb = combB; g.bias = bg; g.V = yaf; g.V2 = ymf;
    gemm_nt_k<EP_GATE><<<dim3(D_/64, BTA_/64, 1), 128, 0, stream>>>(g); }
  { GemmArgs g = G(combB, WoutT, BTA_, D_, D_, D_, D_, D_); g.Cf = combF;
    gemm_nt_k<EP_F32><<<dim3(D_/64, BTA_/64, 1), 128, 0, stream>>>(g); }
  ln_out1_k<<<BT_, 256, 0, stream>>>(combF, x, g_out, b_out, out1);

  // ---- FFN ----
  ln_rows_k<<<BT_, 256, 0, stream>>>(out1, g_ffn, b_ffn, tln);
  { GemmArgs g = G(tln, Wf1T, BT_, FD_, D_, D_, D_, FD_);
    g.Cb = ub; g.bias = bf1;
    gemm_nt_k<EP_BIAS_SILU><<<dim3(FD_/64, BT_/64, 1), 128, 0, stream>>>(g); }
  { GemmArgs g = G(ub, Wf2T, BT_, D_, FD_, FD_, FD_, D_);
    g.Cf = out; g.bias = bf2; g.R = out1;
    gemm_nt_k<EP_RES_BIAS><<<dim3(D_/64, BT_/64, 1), 128, 0, stream>>>(g); }
}
